// AttentionWeigths_15710990369022
// MI455X (gfx1250) — matrix operand order repaired and checked by reading, not compiled, not run on hardware
//
#include <hip/hip_runtime.h>

typedef __attribute__((ext_vector_type(16))) __bf16 v16bf;
typedef __attribute__((ext_vector_type(8)))  float  v8f;
typedef __attribute__((ext_vector_type(8)))  float  vf8;
typedef __attribute__((ext_vector_type(8)))  __bf16 vbf8;
typedef __attribute__((ext_vector_type(4)))  float  vf4e;
typedef __attribute__((ext_vector_type(4)))  __bf16 vbf4;

#define S_LEN  2048
#define D_HEAD 128
#define BM     128    // query rows per workgroup (16 per wave x 8 waves)
#define BN     32     // keys per iteration
#define NWAVES 8

// ---- 16 bf16 fragment from two 16B-aligned 8-element segments -------------
static __device__ inline v16bf ld_frag2(const __bf16* p0, const __bf16* p1) {
  union { uint4 u[2]; v16bf v; } f;
  f.u[0] = *(const uint4*)p0;
  f.u[1] = *(const uint4*)p1;
  return f.v;
}
static __device__ inline v16bf ld_frag16(const __bf16* p) { return ld_frag2(p, p + 8); }

static __device__ inline v8f wmma_bf16(v16bf a, v16bf b, v8f c) {
  return __builtin_amdgcn_wmma_f32_16x16x32_bf16(
      /*neg_a=*/false, a, /*neg_b=*/false, b,
      /*c_mod=*/(short)0, c, /*reuse_a=*/false, /*reuse_b=*/false);
}

// ---- row-max across a 16-lane DPP row: xor1,2,4,8 — pure VALU -------------
#define DPP_MAX_STEP(v, ctrl)                                                 \
  do {                                                                        \
    int _t = __builtin_amdgcn_update_dpp(0, __float_as_int(v), (ctrl),        \
                                         0xf, 0xf, true);                     \
    (v) = fmaxf((v), __int_as_float(_t));                                     \
  } while (0)

static __device__ inline float dppmax16(float v) {
  DPP_MAX_STEP(v, 0xB1);   // quad_perm(1,0,3,2)  : xor 1
  DPP_MAX_STEP(v, 0x4E);   // quad_perm(2,3,0,1)  : xor 2
  DPP_MAX_STEP(v, 0x141);  // row_half_mirror     : combine across quads (8)
  DPP_MAX_STEP(v, 0x140);  // row_mirror          : combine across halves (16)
  return v;
}

__global__ __launch_bounds__(256, 1)
void flash_attn_fwd_bf16wmma(const float* __restrict__ Q,
                             const float* __restrict__ K,
                             const float* __restrict__ V,
                             float* __restrict__ O) {
  // Double-buffered tiles.
  // K tile: row-major bf16, row stride 136 elems (272B, 16B aligned, bank-skewed)
  __shared__ __bf16 sK[2][BN][D_HEAD + 8];
  // V tile transposed: sVT[buf][d][key], row stride 40 elems (80B, 16B aligned)
  __shared__ __bf16 sVT[2][D_HEAD][BN + 8];
  // per-wave P scratch for D-layout -> A-layout transpose
  __shared__ __bf16 sP[NWAVES][16][BN + 8];

  const int tid  = threadIdx.x;
  const int wave = tid >> 5;
  const int lane = tid & 31;
  const int lh   = lane & 15;   // lane within 16-lane half
  const int hi   = lane >> 4;   // which half

  const size_t base = (size_t)blockIdx.y * S_LEN * D_HEAD;   // [b,h] slice
  const int qrow = blockIdx.x * BM + wave * 16 + lh;

  // ---- Q stripe once, native cvt to bf16, A-matrix layout -----------------
  // A 16x32 bf16, k order chosen here (B matches): lanes 0-15 rows M=0..15 hold K {0..15};
  //               lanes 16-31 same rows hold K {16..31}.
  v16bf aq[4];
#pragma unroll
  for (int kc = 0; kc < 4; ++kc) {
    const float* qp = Q + base + (size_t)qrow * D_HEAD + kc * 32 + hi * 16;
    vf8 lo = *(const vf8*)qp;          // d = +0..7
    vf8 hv = *(const vf8*)(qp + 8);    // d = +8..15
    union { vbf8 h[2]; v16bf v; } f;
    f.h[0] = __builtin_convertvector(lo, vbf8);
    f.h[1] = __builtin_convertvector(hv, vbf8);
    aq[kc] = f.v;
  }

  // ones fragment for rowsum-by-WMMA (P x 1s replicates each row's sum)
  union { unsigned u[8]; v16bf v; } onef;
#pragma unroll
  for (int i = 0; i < 8; ++i) onef.u[i] = 0x3F803F80u;  // two bf16 1.0
  const v16bf ones = onef.v;

  const v8f zero8 = {0.f, 0.f, 0.f, 0.f, 0.f, 0.f, 0.f, 0.f};
  v8f acc[8];
#pragma unroll
  for (int dt = 0; dt < 8; ++dt) acc[dt] = zero8;
  float mrow[8], lrow[8];
#pragma unroll
  for (int r = 0; r < 8; ++r) { mrow[r] = -__builtin_inff(); lrow[r] = 0.f; }

  // fold 1/sqrt(128) and log2(e) into one scale; use native exp2
  const float sfac = 0.088388347648318447f * 1.4426950408889634f;

  // staging: 256 threads cover 32x128 tile, 16 floats each
  const int krow = tid >> 3;        // 0..31
  const int kcb  = (tid & 7) * 16;  // 16-col chunk
  const float* kg = K + base + (size_t)krow * D_HEAD + kcb;
  const float* vg = V + base + (size_t)krow * D_HEAD + kcb;

  auto gload = [&](int kt, vf4e* gk, vf4e* gv) {
    const vf4e* kp = (const vf4e*)(kg + (size_t)kt * D_HEAD);
    const vf4e* vp = (const vf4e*)(vg + (size_t)kt * D_HEAD);
#pragma unroll
    for (int j = 0; j < 4; ++j) { gk[j] = kp[j]; gv[j] = vp[j]; }
  };
  auto sstore = [&](int nb, const vf4e* gk, const vf4e* gv) {
    unsigned* kd = (unsigned*)&sK[nb][krow][kcb];
#pragma unroll
    for (int j = 0; j < 4; ++j) {
      union { vbf4 h; unsigned u[2]; } o;
      o.h = __builtin_convertvector(gk[j], vbf4);   // v_cvt_pk_bf16_f32 path
      kd[2 * j] = o.u[0]; kd[2 * j + 1] = o.u[1];
    }
#pragma unroll
    for (int j = 0; j < 4; ++j) {
      vbf4 h = __builtin_convertvector(gv[j], vbf4);
      sVT[nb][kcb + 4 * j + 0][krow] = h.x;
      sVT[nb][kcb + 4 * j + 1][krow] = h.y;
      sVT[nb][kcb + 4 * j + 2][krow] = h.z;
      sVT[nb][kcb + 4 * j + 3][krow] = h.w;
    }
  };

  // prologue: stage tile 0 into buffer 0
  {
    vf4e gk[4], gv[4];
    gload(0, gk, gv);
    sstore(0, gk, gv);
  }
  __syncthreads();

  int p = 0;
  for (int kt = 0; kt < S_LEN; kt += BN) {
    // issue next tile's global loads first; they retire behind 17 WMMAs
    vf4e gk[4], gv[4];
    const bool more = (kt + BN) < S_LEN;
    if (more) {
      gload(kt + BN, gk, gv);
      if (kt + 2 * BN < S_LEN) {  // global_prefetch_b8 for the tile after
        __builtin_prefetch(kg + (size_t)(kt + 2 * BN) * D_HEAD, 0, 0);
        __builtin_prefetch(vg + (size_t)(kt + 2 * BN) * D_HEAD, 0, 0);
      }
    }

    const __bf16(*kb)[D_HEAD + 8] = sK[p];
    const __bf16(*vb)[BN + 8]     = sVT[p];

    // ---- scores: 16x32 per wave; manual 2-stage pipeline on B frags ------
    v8f c0 = zero8, c1 = zero8;
    v16bf b0 = ld_frag16(&kb[lh][hi * 16]);
    v16bf b1 = ld_frag16(&kb[16 + lh][hi * 16]);
#pragma unroll
    for (int kc = 0; kc < 4; ++kc) {
      v16bf n0 = b0, n1 = b1;
      if (kc < 3) {
        n0 = ld_frag16(&kb[lh][(kc + 1) * 32 + hi * 16]);
        n1 = ld_frag16(&kb[16 + lh][(kc + 1) * 32 + hi * 16]);
      }
      c0 = wmma_bf16(aq[kc], b0, c0);
      c1 = wmma_bf16(aq[kc], b1, c1);
      b0 = n0; b1 = n1;
    }

    // ---- online softmax: DPP row-max, exp2, P -> per-wave LDS scratch ----
    float alpha[8];
#pragma unroll
    for (int r = 0; r < 8; ++r) {
      float s0 = c0[r] * sfac;
      float s1 = c1[r] * sfac;
      float mx = dppmax16(fmaxf(s0, s1));
      float mn = fmaxf(mrow[r], mx);
      float a  = __builtin_amdgcn_exp2f(mrow[r] - mn);  // exp2(-inf)=0 first tile
      float p0 = __builtin_amdgcn_exp2f(s0 - mn);
      float p1 = __builtin_amdgcn_exp2f(s1 - mn);
      mrow[r]  = mn;
      alpha[r] = a;
      lrow[r] *= a;
      const int prow = r + hi * 8;
      sP[wave][prow][lh]      = (__bf16)p0;
      sP[wave][prow][16 + lh] = (__bf16)p1;
    }

    // P as A-fragment (intra-wave DS ops are in-order: no barrier needed)
    const __bf16* pp = &sP[wave][lh][hi * 16];
    v16bf ap = ld_frag2(pp, pp + 8);

    // rowsum(P) via WMMA against ones: D-layout replicates sum per row group
    v8f rs = wmma_bf16(ap, ones, zero8);
#pragma unroll
    for (int r = 0; r < 8; ++r) lrow[r] += rs[r];

    // ---- O = O*alpha + P.V over 8 d-tiles, batched B-frag loads ----------
#pragma unroll
    for (int h = 0; h < 2; ++h) {
      v16bf bv[4];
#pragma unroll
      for (int t = 0; t < 4; ++t)
        bv[t] = ld_frag16(&vb[(h * 4 + t) * 16 + lh][hi * 16]);
#pragma unroll
      for (int t = 0; t < 4; ++t) {
        const int dt = h * 4 + t;
#pragma unroll
        for (int r = 0; r < 8; ++r) acc[dt][r] *= alpha[r];
        acc[dt] = wmma_bf16(ap, bv[t], acc[dt]);
      }
    }

    // stage next tile into the other buffer; one barrier per iteration
    if (more) sstore(p ^ 1, gk, gv);
    __syncthreads();
    p ^= 1;
  }

  // ---- epilogue: O /= l, store fp32 ---------------------------------------
#pragma unroll
  for (int r = 0; r < 8; ++r) lrow[r] = 1.0f / lrow[r];
#pragma unroll
  for (int dt = 0; dt < 8; ++dt) {
#pragma unroll
    for (int r = 0; r < 8; ++r) {
      const int orow = blockIdx.x * BM + wave * 16 + r + hi * 8;
      const int ocol = dt * 16 + lh;
      O[base + (size_t)orow * D_HEAD + ocol] = acc[dt][r] * lrow[r];
    }
  }
}

extern "C" void kernel_launch(void* const* d_in, const int* in_sizes, int n_in,
                              void* d_out, int out_size, void* d_ws, size_t ws_size,
                              hipStream_t stream) {
  (void)in_sizes; (void)n_in; (void)out_size; (void)d_ws; (void)ws_size;
  const float* q = (const float*)d_in[0];
  const float* k = (const float*)d_in[1];
  const float* v = (const float*)d_in[2];
  float* o = (float*)d_out;

  dim3 grid(S_LEN / BM, 2 * 16);  // (q-tiles, B*H) = (16, 32)
  dim3 block(256);                // 8 waves of 32
  flash_attn_fwd_bf16wmma<<<grid, block, 0, stream>>>(q, k, v, o);
}
